// NNB_31413390803011
// MI455X (gfx1250) — matrix operand order repaired and checked by reading, not compiled, not run on hardware
//
#include <hip/hip_runtime.h>

// ---------------------------------------------------------------------------
// Problem constants
// ---------------------------------------------------------------------------
static constexpr int BN  = 8192;   // batch
static constexpr int SN  = 100;    // sites per spin
static constexpr int HN  = 512;    // hidden
static constexpr int NF  = 50;     // filled orbitals (NUP == NDN)
static constexpr int KP  = 128;    // padded S (K of GEMM1)
static constexpr int N2  = SN * NF;      // 5000 valid cols of GEMM2
static constexpr int N2P = 5120;         // padded N of GEMM2 (mult of 64)

typedef __attribute__((ext_vector_type(16))) _Float16 v16h;
typedef __attribute__((ext_vector_type(8)))  float    v8f;

// ---------------------------------------------------------------------------
// CDNA5 async global->LDS copy (ASYNCcnt path) via inline asm.
// ISA: GLOBAL_LOAD_ASYNC_TO_LDS_B128 vdst(LDS byte addr), vaddr(64b), off
// LDS byte address == low 32 bits of the generic pointer (flat LDS aperture).
// Host pass never sees the AMD asm (guarded by __AMDGCN__).
// ---------------------------------------------------------------------------
__device__ __forceinline__ void cp16_g2l(const _Float16* g, _Float16* l) {
#if defined(__AMDGCN__)
  unsigned lds_addr = (unsigned)(unsigned long long)(void*)l;
  unsigned long long gaddr = (unsigned long long)(const void*)g;
  asm volatile("global_load_async_to_lds_b128 %0, %1, off"
               :
               : "v"(lds_addr), "v"(gaddr)
               : "memory");
#else
  *(uint4*)l = *(const uint4*)g;
#endif
}

__device__ __forceinline__ void wait_async_all() {
#if defined(__AMDGCN__)
  asm volatile("s_wait_asynccnt 0" ::: "memory");
#endif
}

// ---------------------------------------------------------------------------
// Prep: cfg (0/1 f32) -> f16 padded row (K=128) + rank table
// rank[b][r] = position of site r among filled sites (sorted), -1 if empty.
// ---------------------------------------------------------------------------
__global__ void prep_cfg(const float* __restrict__ cf, _Float16* __restrict__ cfg16,
                         signed char* __restrict__ rank, int off) {
  int b = blockIdx.x * blockDim.x + threadIdx.x;
  if (b >= BN) return;
  const float* row = cf + (size_t)b * (2 * SN) + off;
  int cnt = 0;
  for (int s = 0; s < SN; ++s) {
    float v = row[s];
    cfg16[(size_t)b * KP + s] = (_Float16)v;
    signed char r = -1;
    if (v > 0.5f) { r = (signed char)cnt; ++cnt; }
    rank[(size_t)b * KP + s] = r;
  }
  for (int s = SN; s < KP; ++s) {
    cfg16[(size_t)b * KP + s] = (_Float16)0.f;
    rank[(size_t)b * KP + s] = -1;
  }
}

// W1 (100x512 f32, row-major) -> W1t (512x128 f16, row n contiguous in k, pad 0)
__global__ void prep_w1t(const float* __restrict__ W1, _Float16* __restrict__ W1t) {
  int idx = blockIdx.x * blockDim.x + threadIdx.x;
  if (idx >= HN * KP) return;
  int n = idx / KP, k = idx - n * KP;
  W1t[idx] = (k < SN) ? (_Float16)W1[(size_t)k * HN + n] : (_Float16)0.f;
}

// W2 (512x5000 f32, row-major) -> W2t (5120x512 f16, transposed, pad 0)
__global__ void prep_w2t(const float* __restrict__ W2, _Float16* __restrict__ W2t) {
  int idx = blockIdx.x * blockDim.x + threadIdx.x;
  if (idx >= N2P * HN) return;
  int n = idx / HN, k = idx - n * HN;
  W2t[idx] = (n < N2) ? (_Float16)W2[(size_t)k * N2 + n] : (_Float16)0.f;
}

// ---------------------------------------------------------------------------
// WMMA GEMM, 128x64 macro-tile, 128 threads (4 waves), double-buffered LDS
// fed by async global->LDS copies. C = A(MxK f16 rm) @ Bt(NxK f16 rm)^T.
// Wave w owns C rows [32w, 32w+32): 2 A-frags x 4 B-frags = 8 WMMA / k-step.
// MODE 0: outH = f16( relu(C + bias) )                (GEMM1 -> h)
// MODE 1: scatter C + bias into mats via rank table   (GEMM2 -> 50x50 mats)
// ---------------------------------------------------------------------------
template <int MODE>
__global__ __launch_bounds__(128)
void wmma_gemm(const _Float16* __restrict__ A, const _Float16* __restrict__ Bt,
               int K, int N, const float* __restrict__ bias,
               _Float16* __restrict__ outH,
               float* __restrict__ mats,
               const signed char* __restrict__ rank,
               int nValid) {
  // LDS row stride 40 halves = 80B: 16B-aligned b128, conflict-free frag reads
  constexpr int LDT = 40;
  __shared__ _Float16 As[2][128 * LDT];
  __shared__ _Float16 Bs[2][64 * LDT];

  const int t    = threadIdx.x;
  const int lane = t & 31;
  const int w    = t >> 5;               // wave 0..3
  const int m0   = blockIdx.y * 128;
  const int n0   = blockIdx.x * 64;

  v8f acc[2][4];
  #pragma unroll
  for (int r = 0; r < 2; ++r)
    #pragma unroll
    for (int i = 0; i < 4; ++i)
      #pragma unroll
      for (int e = 0; e < 8; ++e) acc[r][i][e] = 0.f;

  const int steps = K >> 5;

  auto load_stage = [&](int s, int buf) {
    const int kb = s * 32;
    #pragma unroll
    for (int i = 0; i < 4; ++i) {        // A tile: 128 rows x 4 chunks = 512
      int c = t + i * 128;
      int row = c >> 2, seg = c & 3;
      cp16_g2l(A + (size_t)(m0 + row) * K + kb + seg * 8,
               &As[buf][row * LDT + seg * 8]);
    }
    #pragma unroll
    for (int i = 0; i < 2; ++i) {        // B tile: 64 rows x 4 chunks = 256
      int c = t + i * 128;
      int row = c >> 2, seg = c & 3;
      cp16_g2l(Bt + (size_t)(n0 + row) * K + kb + seg * 8,
               &Bs[buf][row * LDT + seg * 8]);
    }
  };

  load_stage(0, 0);
  wait_async_all();
  __syncthreads();

  const int mr  = lane & 15;
  const int ah0 = (lane < 16) ? 0 : 8;    // A frag: k {0-7,16-23} / {8-15,24-31}
  const int bk0 = (lane < 16) ? 0 : 8;    // B frag: k {0-7,16-23} / {8-15,24-31}

  for (int s = 0; s < steps; ++s) {
    const int buf = s & 1;
    if (s + 1 < steps) load_stage(s + 1, buf ^ 1);  // prefetch next stage

    v16h a0, a1;
    {
      union { v16h h; uint4 u[2]; } f;
      const _Float16* p = &As[buf][(w * 32 + mr) * LDT];
      f.u[0] = *(const uint4*)(p + ah0);
      f.u[1] = *(const uint4*)(p + ah0 + 16);
      a0 = f.h;
      const _Float16* q = p + 16 * LDT;
      f.u[0] = *(const uint4*)(q + ah0);
      f.u[1] = *(const uint4*)(q + ah0 + 16);
      a1 = f.h;
    }
    #pragma unroll
    for (int nf = 0; nf < 4; ++nf) {
      union { v16h h; uint4 u[2]; } g;
      const _Float16* p = &Bs[buf][(nf * 16 + mr) * LDT + bk0];
      g.u[0] = *(const uint4*)(p);
      g.u[1] = *(const uint4*)(p + 16);
      acc[0][nf] = __builtin_amdgcn_wmma_f32_16x16x32_f16(
          false, a0, false, g.h, (short)0, acc[0][nf], false, false);
      acc[1][nf] = __builtin_amdgcn_wmma_f32_16x16x32_f16(
          false, a1, false, g.h, (short)0, acc[1][nf], false, false);
    }

    if (s + 1 < steps) {
      wait_async_all();    // stage s+1 landed (own wave)
      __syncthreads();     // all waves landed + done reading buf before reuse
    }
  }

  // Epilogue. C layout: lane -> n = lane%16 ; vgpr v -> m = v + (lane<16?0:8)
  const int col = lane & 15;
  const int rb  = (lane < 16) ? 0 : 8;
  #pragma unroll
  for (int rblk = 0; rblk < 2; ++rblk) {
    #pragma unroll
    for (int nf = 0; nf < 4; ++nf) {
      int n = n0 + nf * 16 + col;
      float bv = (n < nValid) ? bias[n] : 0.f;
      #pragma unroll
      for (int v = 0; v < 8; ++v) {
        int m = m0 + w * 32 + rblk * 16 + rb + v;
        float val = acc[rblk][nf][v] + bv;
        if (MODE == 0) {
          val = val > 0.f ? val : 0.f;
          outH[(size_t)m * N + n] = (_Float16)val;
        } else {
          if (n < nValid) {
            int r = n / NF;
            int j = n - r * NF;
            int i = rank[(size_t)m * KP + r];
            if (i >= 0) mats[(size_t)m * (NF * NF) + i * NF + j] = val;
          }
        }
      }
    }
  }
}

// ---------------------------------------------------------------------------
// 50x50 LU with partial pivoting in LDS; one workgroup (64 threads) per matrix.
// det accumulated in double (magnitudes ~1e-19 per spin).
// ---------------------------------------------------------------------------
__global__ __launch_bounds__(64)
void det_lu(const float* __restrict__ mats, double* __restrict__ dets) {
  __shared__ float M[NF * 52];
  __shared__ int s_piv;
  const int t = threadIdx.x;
  const int b = blockIdx.x;
  const float* src = mats + (size_t)b * (NF * NF);
  for (int idx = t; idx < NF * NF; idx += 64) {
    int r = idx / NF, c = idx - r * NF;
    M[r * 52 + c] = src[idx];
  }
  __syncthreads();
  int sgn = 1;
  for (int k = 0; k < NF; ++k) {
    if (t == 0) {
      int p = k; float best = fabsf(M[k * 52 + k]);
      for (int i = k + 1; i < NF; ++i) {
        float a = fabsf(M[i * 52 + k]);
        if (a > best) { best = a; p = i; }
      }
      s_piv = p;
    }
    __syncthreads();
    int p = s_piv;
    if (p != k) {
      sgn = -sgn;
      for (int j = t; j < NF; j += 64) {
        float tmp = M[k * 52 + j]; M[k * 52 + j] = M[p * 52 + j]; M[p * 52 + j] = tmp;
      }
    }
    __syncthreads();
    float piv = M[k * 52 + k];
    float inv = (piv != 0.f) ? (1.f / piv) : 0.f;
    for (int i = k + 1 + t; i < NF; i += 64) {
      float l = M[i * 52 + k] * inv;
      for (int j = k + 1; j < NF; ++j) M[i * 52 + j] -= l * M[k * 52 + j];
    }
    __syncthreads();
  }
  if (t == 0) {
    double pr = (double)sgn;
    for (int k = 0; k < NF; ++k) pr *= (double)M[k * 52 + k];
    dets[b] = pr;
  }
}

__global__ void combine(const double* __restrict__ du, const double* __restrict__ dd,
                        float* __restrict__ out) {
  int b = blockIdx.x * blockDim.x + threadIdx.x;
  if (b < BN) out[b] = (float)(du[b] * dd[b]);
}

// ---------------------------------------------------------------------------
// Host driver
// ---------------------------------------------------------------------------
extern "C" void kernel_launch(void* const* d_in, const int* in_sizes, int n_in,
                              void* d_out, int out_size, void* d_ws, size_t ws_size,
                              hipStream_t stream) {
  const float* cf = (const float*)d_in[0];

  char* wsb = (char*)d_ws;
  size_t off = 0;
  auto carve = [&](size_t bytes) {
    void* p = wsb + off;
    off = (off + bytes + 255) & ~(size_t)255;
    return p;
  };
  _Float16* cfg16 = (_Float16*)carve((size_t)BN * KP * 2);       //  2 MB (reused per spin)
  signed char* rank = (signed char*)carve((size_t)BN * KP);      //  1 MB (reused)
  _Float16* W1t  = (_Float16*)carve((size_t)HN * KP * 2);        //  128 KB (reused)
  _Float16* W2t  = (_Float16*)carve((size_t)N2P * HN * 2);       //  5 MB (reused)
  _Float16* h16  = (_Float16*)carve((size_t)BN * HN * 2);        //  8 MB (reused)
  float*    mats = (float*)carve((size_t)BN * NF * NF * 4);      // 82 MB (reused)
  double*   detU = (double*)carve((size_t)BN * 8);
  double*   detD = (double*)carve((size_t)BN * 8);
  // total ~99 MB

  for (int spin = 0; spin < 2; ++spin) {
    const float* W1 = (const float*)d_in[1 + 4 * spin];
    const float* b1 = (const float*)d_in[2 + 4 * spin];
    const float* W2 = (const float*)d_in[3 + 4 * spin];
    const float* b2 = (const float*)d_in[4 + 4 * spin];
    double* det = spin ? detD : detU;
    int cfoff = spin ? SN : 0;

    prep_cfg<<<(BN + 255) / 256, 256, 0, stream>>>(cf, cfg16, rank, cfoff);
    prep_w1t<<<(HN * KP + 255) / 256, 256, 0, stream>>>(W1, W1t);
    prep_w2t<<<(N2P * HN + 255) / 256, 256, 0, stream>>>(W2, W2t);

    // GEMM1: h = relu(cfg @ W1 + b1)   (M=8192, K=128, N=512)
    wmma_gemm<0><<<dim3(HN / 64, BN / 128), 128, 0, stream>>>(
        cfg16, W1t, KP, HN, b1, h16, nullptr, nullptr, HN);

    // GEMM2 + gather: mats = (h @ W2 + b2)[filled rows]  (M=8192, K=512, N=5120)
    wmma_gemm<1><<<dim3(N2P / 64, BN / 128), 128, 0, stream>>>(
        h16, W2t, HN, N2P, b2, nullptr, mats, rank, N2);

    det_lu<<<BN, 64, 0, stream>>>(mats, det);
  }

  combine<<<(BN + 255) / 256, 256, 0, stream>>>(detU, detD, (float*)d_out);
}
